// TransVLAD3d_82128364634646
// MI455X (gfx1250) — matrix operand order repaired and checked by reading, not compiled, not run on hardware
//
#include <hip/hip_runtime.h>
#include <hip/hip_bf16.h>

#define N_    16
#define C_    128
#define T_    16384
#define K_    64
#define KG_   73
#define KGP_  80
#define SPLIT_ 32
#define CHUNK_ (T_ / SPLIT_)   // 512 columns per block
#define TILE_  64              // columns per inner tile
#define NTILES_ (CHUNK_ / TILE_)
#define EPSF  1e-12f
#define UNI_BYTES 18688        // max( x_t 64*128*2 = 16384 , l_lds 73*64*4 = 18688 )

typedef __attribute__((ext_vector_type(16))) __bf16       v16bf;
typedef __attribute__((ext_vector_type(8)))  float        v8f;
typedef __attribute__((ext_vector_type(4)))  unsigned int v4u;
typedef unsigned short ushort_t;

// ---- bf16 helpers: native hardware converts (v_cvt), bit-punned to ushort ----
__device__ inline ushort_t f2bf(float x) {
    union { __bf16 b; ushort_t u; } v; v.b = (__bf16)x; return v.u;
}
__device__ inline float bf2f(ushort_t h) {
    union { ushort_t u; __bf16 b; } v; v.u = h; return (float)v.b;
}

union F16x16 { v16bf v; v4u q[2]; };
union PK8    { ushort_t h[8]; v4u q; };

// ---- WMMA fragment loaders: contraction dim contiguous -> 2x ds_load_b128 ----
// A 16x32 bf16, storage p[row*ld + k] (K contiguous).
// lanes 0-15: M=lane, K {0..7,16..23}; lanes 16-31: M=lane-16, K {8..15,24..31}
__device__ inline v16bf ldfragA(const ushort_t* p, int ld, int lane) {
    int m   = lane & 15;
    int klo = (lane & 16) ? 8 : 0;
    const ushort_t* r = p + m * ld + klo;
    F16x16 u;
    u.q[0] = *(const v4u*)(r);        // K elems klo..klo+7
    u.q[1] = *(const v4u*)(r + 16);   // K elems 16+klo..16+klo+7
    return u.v;
}
// B 32x16 bf16 where storage is transposed: B[k][n] = p[n*ld + k] (K contiguous).
// lanes 0-15: col=lane, K {0..7,16..23}; lanes 16-31: col=lane-16, K {8..15,24..31}
__device__ inline v16bf ldfragBT(const ushort_t* p, int ld, int lane) {
    int n   = lane & 15;
    int klo = (lane & 16) ? 8 : 0;
    const ushort_t* r = p + n * ld + klo;
    F16x16 u;
    u.q[0] = *(const v4u*)(r);        // K elems klo..klo+7
    u.q[1] = *(const v4u*)(r + 16);   // K elems 16+klo..16+klo+7
    return u.v;
}

__device__ inline v8f wmma_bf16(v16bf a, v16bf b, v8f c) {
    return __builtin_amdgcn_wmma_f32_16x16x32_bf16(false, a, false, b, (short)0, c, false, false);
}

// one GEMM1 output tile: W[mt] (16x128) * xn (128x16 cols of nt), K-loop over c
__device__ inline v8f gemm1_tile(const ushort_t* w, const ushort_t* xt,
                                 int mt, int nt, int lane) {
    v8f c = (v8f){0.f,0.f,0.f,0.f,0.f,0.f,0.f,0.f};
#pragma unroll
    for (int ks = 0; ks < 4; ++ks) {
        v16bf a = ldfragA (w  + (mt * 16) * C_ + ks * 32, C_, lane);
        v16bf b = ldfragBT(xt + (nt * 16) * C_ + ks * 32, C_, lane);
        c = wmma_bf16(a, b, c);
    }
    return c;
}
__device__ inline void store_logits(float* l, v8f c, int mt, int nt, int lane) {
    int mb  = mt * 16 + ((lane & 16) ? 8 : 0);
    int col = nt * 16 + (lane & 15);
#pragma unroll
    for (int r = 0; r < 8; ++r) {
        int m = mb + r;
        if (m < KG_) l[m * TILE_ + col] = c[r];
    }
}

// ---------------- workspace zero ----------------
__global__ void ws_zero_kernel(float* p, int n) {
    int i = blockIdx.x * blockDim.x + threadIdx.x;
    if (i < n) p[i] = 0.0f;
}

// ---------------- fused main kernel ----------------
// grid: N_*SPLIT_ blocks, 256 threads (8 waves)
__global__ __launch_bounds__(256) void vlad_main_kernel(
        const float* __restrict__ x,        // [N,C,T]
        const float* __restrict__ conv_w,   // [KG,C]
        float* __restrict__ ws_vlad,        // [N,K,C] accumulators
        float* __restrict__ ws_asum)        // [N,K]
{
    __shared__ __attribute__((aligned(16))) ushort_t w_lds[KGP_ * C_];  // W bf16 [80][128]
    __shared__ __attribute__((aligned(16))) unsigned char uni_lds[UNI_BYTES];
    __shared__ __attribute__((aligned(16))) ushort_t x_c[C_ * TILE_];   // xn bf16 [128][64] (t contig)
    __shared__ __attribute__((aligned(16))) ushort_t s_lds[K_ * TILE_]; // soft bf16 [64][64] (t contig)
    __shared__ float csum[4 * TILE_];

    ushort_t* x_t   = (ushort_t*)uni_lds;   // xn bf16 [64][128] (c contig) -- GEMM1 phase
    float*    l_lds = (float*)uni_lds;      // logits f32 [73][64]          -- softmax phase

    const int tid  = threadIdx.x;
    const int lane = tid & 31;
    const int wv   = tid >> 5;
    const int n     = blockIdx.x / SPLIT_;
    const int chunk = blockIdx.x % SPLIT_;
    const int tbase = chunk * CHUNK_;
    const float* xb = x + (size_t)n * C_ * T_;

    // conv_w -> bf16 LDS (rows [73,80) zero), vectorized b128 stores
    for (int i8 = tid; i8 < (KGP_ * C_) / 8; i8 += 256) {
        int base = i8 * 8;
        int r    = base >> 7;
        PK8 pk;
        if (r < KG_) {
            const float* src = conv_w + base;
#pragma unroll
            for (int j = 0; j < 8; ++j) pk.h[j] = f2bf(src[j]);
        } else {
            pk.q = (v4u){0u, 0u, 0u, 0u};
        }
        *(v4u*)&w_lds[base] = pk.q;
    }

    // persistent GEMM2 accumulators: wave wv owns Mtile=wv>>1, Ntiles (wv&1)*4..+3
    v8f acc0 = (v8f){0.f,0.f,0.f,0.f,0.f,0.f,0.f,0.f};
    v8f acc1 = acc0, acc2 = acc0, acc3 = acc0;
    float asum_reg = 0.0f;
    const int m2  = wv >> 1;
    const int nb2 = (wv & 1) * 4;

    __syncthreads();

    for (int tile = 0; tile < NTILES_; ++tile) {
        const int t0  = tbase + tile * TILE_;
        const int col = tid & 63;
        const int grp = tid >> 6;   // 0..3, 32 channel rows each
        const float* xp = xb + (size_t)(grp * 32) * T_ + t0 + col;

        // ---- pass 1: column sum of squares (x read #1) ----
        {
            float s = 0.f;
#pragma unroll 8
            for (int r = 0; r < 32; ++r) {
                float v = xp[(size_t)r * T_];
                s += v * v;
            }
            csum[grp * 64 + col] = s;
            if (tile + 1 < NTILES_)
                __builtin_prefetch(xp + TILE_, 0, 1);   // global_prefetch_b8
        }
        __syncthreads();
        if (tid < 64) {
            float s = csum[tid] + csum[64 + tid] + csum[128 + tid] + csum[192 + tid];
            csum[tid] = 1.0f / fmaxf(sqrtf(s), EPSF);
        }
        __syncthreads();

        // ---- pass 2: normalize + convert; dual-layout LDS (x read #2, L2-hot) ----
        {
            float inv = csum[col];
            ushort_t* xtrow = x_t + col * C_ + grp * 32;   // contiguous c
#pragma unroll
            for (int r = 0; r < 32; r += 2) {
                float a  = xp[(size_t)r       * T_] * inv;
                float b2 = xp[(size_t)(r + 1) * T_] * inv;
                ushort_t ha = f2bf(a), hb = f2bf(b2);
                *(unsigned*)&xtrow[r] = (unsigned)ha | ((unsigned)hb << 16);
                x_c[(grp * 32 + r)     * TILE_ + col] = ha;
                x_c[(grp * 32 + r + 1) * TILE_ + col] = hb;
            }
        }
        __syncthreads();

        // ---- GEMM1: logits[80x64] = W * xn ; fragments kept in regs ----
        {
            const int nt  = wv & 3;
            const int mt0 = (wv < 4) ? 0 : 3;
            v8f f0 = gemm1_tile(w_lds, x_t, mt0,     nt, lane);
            v8f f1 = gemm1_tile(w_lds, x_t, mt0 + 1, nt, lane);
            v8f f2 = (v8f){0.f,0.f,0.f,0.f,0.f,0.f,0.f,0.f};
            if (wv < 4) f2 = gemm1_tile(w_lds, x_t, 2, nt, lane);  // wave-uniform branch
            __syncthreads();              // all x_t reads done; region becomes l_lds
            store_logits(l_lds, f0, mt0,     nt, lane);
            store_logits(l_lds, f1, mt0 + 1, nt, lane);
            if (wv < 4) store_logits(l_lds, f2, 2, nt, lane);
        }
        __syncthreads();

        // ---- softmax over 73 per column; keep 64 as bf16 ----
        if (tid < TILE_) {
            int t = tid;
            float mx = -3.4e38f;
            for (int k = 0; k < KG_; ++k) mx = fmaxf(mx, l_lds[k * TILE_ + t]);
            float den = 0.f;
            for (int k = 0; k < KG_; ++k) {
                float e = __expf(l_lds[k * TILE_ + t] - mx);
                l_lds[k * TILE_ + t] = e;
                den += e;
            }
            float inv = 1.0f / den;
            for (int k = 0; k < K_; ++k)
                s_lds[k * TILE_ + t] = f2bf(l_lds[k * TILE_ + t] * inv);
        }
        __syncthreads();

        // a_sum: vectorized b128 row reads, unpack two bf16 per dword
        if (tid < K_) {
            const v4u* row = (const v4u*)&s_lds[tid * TILE_];
            float s = 0.f;
#pragma unroll
            for (int q8 = 0; q8 < TILE_ / 8; ++q8) {
                v4u q = row[q8];
#pragma unroll
                for (int e = 0; e < 4; ++e) {
                    unsigned u = q[e];
                    union { unsigned u; float f; } lo, hi;
                    lo.u = u << 16;
                    hi.u = u & 0xFFFF0000u;
                    s += lo.f + hi.f;
                }
            }
            asum_reg += s;
        }

        // ---- GEMM2: vlad[64x128] += soft * xn^T ----
#pragma unroll
        for (int ks = 0; ks < 2; ++ks) {
            v16bf a = ldfragA(&s_lds[(m2 * 16) * TILE_ + ks * 32], TILE_, lane);
            v16bf b0 = ldfragBT(&x_c[((nb2 + 0) * 16) * TILE_ + ks * 32], TILE_, lane);
            acc0 = wmma_bf16(a, b0, acc0);
            v16bf b1 = ldfragBT(&x_c[((nb2 + 1) * 16) * TILE_ + ks * 32], TILE_, lane);
            acc1 = wmma_bf16(a, b1, acc1);
            v16bf b2 = ldfragBT(&x_c[((nb2 + 2) * 16) * TILE_ + ks * 32], TILE_, lane);
            acc2 = wmma_bf16(a, b2, acc2);
            v16bf b3 = ldfragBT(&x_c[((nb2 + 3) * 16) * TILE_ + ks * 32], TILE_, lane);
            acc3 = wmma_bf16(a, b3, acc3);
        }
        __syncthreads();
    }

    // ---- flush partials ----
    {
        float* dst = ws_vlad + (size_t)n * K_ * C_;
        int mbase = m2 * 16 + ((lane & 16) ? 8 : 0);
        int col0  = lane & 15;
        v8f a4[4] = {acc0, acc1, acc2, acc3};
#pragma unroll
        for (int f = 0; f < 4; ++f) {
            int cidx = (nb2 + f) * 16 + col0;
#pragma unroll
            for (int r = 0; r < 8; ++r)
                atomicAdd(&dst[(mbase + r) * C_ + cidx], a4[f][r]);
        }
    }
    if (tid < K_) atomicAdd(&ws_asum[n * K_ + tid], asum_reg);
}

// ---------------- finalize ----------------
__global__ __launch_bounds__(256) void vlad_finalize_kernel(
        const float* __restrict__ ws_vlad,
        const float* __restrict__ ws_asum,
        const float* __restrict__ centroids,   // [K,C]
        const float* __restrict__ cweights,    // [K]
        float* __restrict__ out)               // [N, K*C]
{
    __shared__ float red[256];
    __shared__ float invk[K_];
    __shared__ float asv[K_];
    __shared__ float cwn;

    const int tid = threadIdx.x;
    const int n   = blockIdx.x;

    if (tid < K_) {
        asv[tid] = ws_asum[n * K_ + tid];
        red[tid] = cweights[tid] * cweights[tid];
    }
    __syncthreads();
    if (tid == 0) {
        float s = 0.f;
        for (int k = 0; k < K_; ++k) s += red[k];
        cwn = fmaxf(sqrtf(s), EPSF);
    }
    __syncthreads();

    const int k  = tid >> 2;
    const int c0 = (tid & 3) * 32;
    const float* vp = ws_vlad + (size_t)n * K_ * C_ + k * C_;
    const float* cp = centroids + k * C_;
    const float ak  = asv[k];

    float v[32];
    float ss = 0.f;
#pragma unroll 8
    for (int i = 0; i < 32; ++i) {
        float t = vp[c0 + i] - ak * cp[c0 + i];
        v[i] = t;
        ss += t * t;
    }
    red[tid] = ss;
    __syncthreads();
    if (tid < K_) {
        float s = red[tid * 4] + red[tid * 4 + 1] + red[tid * 4 + 2] + red[tid * 4 + 3];
        float inv = 1.0f / fmaxf(sqrtf(s), EPSF);
        invk[tid] = inv * (cweights[tid] / cwn);
    }
    __syncthreads();

    const float scale = invk[k];
    float gs = 0.f;
#pragma unroll 8
    for (int i = 0; i < 32; ++i) { v[i] *= scale; gs += v[i] * v[i]; }
    red[tid] = gs;
    __syncthreads();
    for (int s = 128; s > 0; s >>= 1) {
        if (tid < s) red[tid] += red[tid + s];
        __syncthreads();
    }
    const float ginv = 1.0f / fmaxf(sqrtf(red[0]), EPSF);
    float* op = out + (size_t)n * K_ * C_ + k * C_;
#pragma unroll 8
    for (int i = 0; i < 32; ++i) op[c0 + i] = v[i] * ginv;
}

extern "C" void kernel_launch(void* const* d_in, const int* in_sizes, int n_in,
                              void* d_out, int out_size, void* d_ws, size_t ws_size,
                              hipStream_t stream) {
    const float* x         = (const float*)d_in[0];
    const float* conv_w    = (const float*)d_in[1];
    const float* centroids = (const float*)d_in[2];
    const float* cweights  = (const float*)d_in[3];

    float* ws_vlad = (float*)d_ws;
    float* ws_asum = ws_vlad + (size_t)N_ * K_ * C_;

    const int zn = N_ * K_ * C_ + N_ * K_;
    ws_zero_kernel<<<(zn + 255) / 256, 256, 0, stream>>>(ws_vlad, zn);
    vlad_main_kernel<<<N_ * SPLIT_, 256, 0, stream>>>(x, conv_w, ws_vlad, ws_asum);
    vlad_finalize_kernel<<<N_, 256, 0, stream>>>(ws_vlad, ws_asum, centroids, cweights,
                                                 (float*)d_out);
}
